// BulkRNAEncoder_30056181137753
// MI455X (gfx1250) — compile-verified, matrix operand order checked by reading, not run on hardware
//
#include <hip/hip_runtime.h>
#include <hip/hip_bf16.h>
#include <math.h>

typedef __attribute__((ext_vector_type(16))) __bf16 v16bf;
typedef __attribute__((ext_vector_type(8)))  float  v8f;

#define NB 32
#define KG 2000
#define DM 256
#define HH 4
#define DH 64
#define LLAYERS 2
#define SSQ (KG + 1)          // 2001 tokens
#define NSROWS (NB * SSQ)     // 64032
#define MF 266                // performer features
#define MP 272                // padded to 16
#define DFF 1024

__device__ __forceinline__ unsigned short f2bf(float f) {
  unsigned int x = __float_as_uint(f);
  x += 0x7FFFu + ((x >> 16) & 1u);   // round-to-nearest-even
  return (unsigned short)(x >> 16);
}
__device__ __forceinline__ unsigned int pack2bf(float a, float b) {
  return (unsigned int)f2bf(a) | ((unsigned int)f2bf(b) << 16);
}
__device__ __forceinline__ float gelu_exact(float x) {
  return 0.5f * x * (1.0f + erff(x * 0.70710678118654752f));
}
__device__ __forceinline__ float waveRedSum(float v) {
#pragma unroll
  for (int m = 16; m > 0; m >>= 1) v += __shfl_xor(v, m, 32);
  return v;
}
__device__ __forceinline__ float waveRedMax(float v) {
#pragma unroll
  for (int m = 16; m > 0; m >>= 1) v = fmaxf(v, __shfl_xor(v, m, 32));
  return v;
}

// ---------------- Generic batched WMMA GEMM ----------------
// C[z] = act( alpha*op(A[z]) @ op(B[z]) + bias ) * rowScale, optionally +=
// batch offset(z) = (z/batchDiv)*s*1 + (z%batchDiv)*s*2
struct GemmArgs {
  const float* A; const float* B; const float* bias; float* C;
  const float* rowScale;
  long long lda, ldb, ldc;
  long long sA1, sA2, sB1, sB2, sC1, sC2, sRS;
  int M, N, K, batchDiv, transA, transB, act, accum;
  float alpha;
};

// 64x64 macro tile, TK=32. 8 waves; wave w owns a 32x16 slab:
//   col tile wc = w & 3, row tiles {2*(w>>2), 2*(w>>2)+1}  -> 2 WMMAs / wave / K-step
__global__ __launch_bounds__(256) void k_gemm(GemmArgs p) {
  const int z  = blockIdx.z;
  const int zq = z / p.batchDiv, zr = z % p.batchDiv;
  const float* A = p.A + zq * p.sA1 + zr * p.sA2;
  const float* B = p.B + zq * p.sB1 + zr * p.sB2;
  float*       C = p.C + zq * p.sC1 + zr * p.sC2;

  const int bm = blockIdx.y * 64;
  const int bn = blockIdx.x * 64;
  const int tid  = threadIdx.x;
  const int wave = tid >> 5, lane = tid & 31;
  const int wc  = wave & 3;
  const int wrb = (wave >> 2) * 2;             // row-tile base (0 or 2)
  const int g   = lane >> 4;
  const int mrow0 = wrb * 16 + (lane & 15);
  const int ncol  = wc * 16 + (lane & 15);

  __shared__ unsigned short Als[64][34];       // bf16 A tile, row-major [m][k]
  __shared__ unsigned short Bls[64][34];       // bf16 B tile, transposed [n][k]

  v8f cacc0 = {0.f,0.f,0.f,0.f,0.f,0.f,0.f,0.f};
  v8f cacc1 = {0.f,0.f,0.f,0.f,0.f,0.f,0.f,0.f};

  for (int k0 = 0; k0 < p.K; k0 += 32) {
    __syncthreads();
    const bool full = (bm + 64 <= p.M) && (bn + 64 <= p.N) && (k0 + 32 <= p.K);
    if (full) {
      // ---- fast path: unguarded float4 loads, block-uniform ----
      if (!p.transA) {
        const int r  = tid >> 2;               // 0..63
        const int c0 = (tid & 3) * 8;          // 0,8,16,24
        const float* src = A + (long long)(bm + r) * p.lda + k0 + c0;
        float4 a0 = *(const float4*)(src);
        float4 a1 = *(const float4*)(src + 4);
        unsigned int* d = (unsigned int*)&Als[r][c0];
        d[0] = pack2bf(a0.x * p.alpha, a0.y * p.alpha);
        d[1] = pack2bf(a0.z * p.alpha, a0.w * p.alpha);
        d[2] = pack2bf(a1.x * p.alpha, a1.y * p.alpha);
        d[3] = pack2bf(a1.z * p.alpha, a1.w * p.alpha);
      } else {
        const int c  = tid >> 3;               // 0..31 (k index)
        const int r0 = (tid & 7) * 8;          // 0..56 (m index)
        const float* src = A + (long long)(k0 + c) * p.lda + bm + r0;
        float4 a0 = *(const float4*)(src);
        float4 a1 = *(const float4*)(src + 4);
        Als[r0+0][c] = f2bf(a0.x * p.alpha); Als[r0+1][c] = f2bf(a0.y * p.alpha);
        Als[r0+2][c] = f2bf(a0.z * p.alpha); Als[r0+3][c] = f2bf(a0.w * p.alpha);
        Als[r0+4][c] = f2bf(a1.x * p.alpha); Als[r0+5][c] = f2bf(a1.y * p.alpha);
        Als[r0+6][c] = f2bf(a1.z * p.alpha); Als[r0+7][c] = f2bf(a1.w * p.alpha);
      }
      if (!p.transB) {
        const int c  = tid >> 3;               // k index
        const int n0 = (tid & 7) * 8;          // n index
        const float* src = B + (long long)(k0 + c) * p.ldb + bn + n0;
        float4 b0 = *(const float4*)(src);
        float4 b1 = *(const float4*)(src + 4);
        Bls[n0+0][c] = f2bf(b0.x); Bls[n0+1][c] = f2bf(b0.y);
        Bls[n0+2][c] = f2bf(b0.z); Bls[n0+3][c] = f2bf(b0.w);
        Bls[n0+4][c] = f2bf(b1.x); Bls[n0+5][c] = f2bf(b1.y);
        Bls[n0+6][c] = f2bf(b1.z); Bls[n0+7][c] = f2bf(b1.w);
      } else {
        const int n  = tid >> 2;
        const int c0 = (tid & 3) * 8;
        const float* src = B + (long long)(bn + n) * p.ldb + k0 + c0;
        float4 b0 = *(const float4*)(src);
        float4 b1 = *(const float4*)(src + 4);
        unsigned int* d = (unsigned int*)&Bls[n][c0];
        d[0] = pack2bf(b0.x, b0.y);
        d[1] = pack2bf(b0.z, b0.w);
        d[2] = pack2bf(b1.x, b1.y);
        d[3] = pack2bf(b1.z, b1.w);
      }
    } else {
      // ---- slow path: guarded scalar loads (boundary tiles only) ----
#pragma unroll
      for (int i = 0; i < 8; ++i) {            // 64x32 A elements
        int e = tid + i * 256;
        int r = e >> 5, c = e & 31;
        int gr = bm + r, gc = k0 + c;
        float v = 0.f;
        if (gr < p.M && gc < p.K)
          v = p.transA ? A[(long long)gc * p.lda + gr]
                       : A[(long long)gr * p.lda + gc];
        Als[r][c] = f2bf(v * p.alpha);
      }
#pragma unroll
      for (int i = 0; i < 8; ++i) {            // 64x32 B elements
        int e = tid + i * 256;
        int n = e >> 5, c = e & 31;
        int gn = bn + n, gc = k0 + c;
        float v = 0.f;
        if (gn < p.N && gc < p.K)
          v = p.transB ? B[(long long)gn * p.ldb + gc]
                       : B[(long long)gc * p.ldb + gn];
        Bls[n][c] = f2bf(v);
      }
    }
    __syncthreads();

    union { v16bf v; unsigned int u[8]; } fa0, fa1, fb;
#pragma unroll
    for (int pp = 0; pp < 8; ++pp) {
      int kk = 2 * pp + 8 * g + ((pp >= 4) ? 8 : 0);   // ISA 16-bit operand layout
      fa0.u[pp] = *(const unsigned int*)&Als[mrow0][kk];
      fa1.u[pp] = *(const unsigned int*)&Als[mrow0 + 16][kk];
      fb.u[pp]  = *(const unsigned int*)&Bls[ncol][kk];
    }
    cacc0 = __builtin_amdgcn_wmma_f32_16x16x32_bf16(
        false, fa0.v, false, fb.v, (short)0, cacc0, false, false);
    cacc1 = __builtin_amdgcn_wmma_f32_16x16x32_bf16(
        false, fa1.v, false, fb.v, (short)0, cacc1, false, false);
  }

  const int gcol = bn + ncol;
  if (gcol < p.N) {
    const float bv = p.bias ? p.bias[gcol] : 0.f;
#pragma unroll
    for (int r = 0; r < 8; ++r) {              // C layout: VGPR r -> M = r + 8*g
      int grow0 = bm + wrb * 16 + r + 8 * g;
      int grow1 = grow0 + 16;
      if (grow0 < p.M) {
        float v = cacc0[r] + bv;
        if (p.act == 1) v = gelu_exact(v);
        if (p.rowScale) v *= p.rowScale[(long long)z * p.sRS + grow0];
        long long idx = (long long)grow0 * p.ldc + gcol;
        if (p.accum) C[idx] += v; else C[idx] = v;
      }
      if (grow1 < p.M) {
        float v = cacc1[r] + bv;
        if (p.act == 1) v = gelu_exact(v);
        if (p.rowScale) v *= p.rowScale[(long long)z * p.sRS + grow1];
        long long idx = (long long)grow1 * p.ldc + gcol;
        if (p.accum) C[idx] += v; else C[idx] = v;
      }
    }
  }
}

// ---------------- Embedding: tok = [cls ; gene+pos+gelu(LN(e*w+b))] ----------
__global__ __launch_bounds__(256) void k_embed(
    const float* __restrict__ expr, const float* __restrict__ gene_emb,
    const float* __restrict__ pos_emb, const float* __restrict__ val_w,
    const float* __restrict__ val_b, const float* __restrict__ lg,
    const float* __restrict__ lb, const float* __restrict__ cls, float* tok) {
  __shared__ float sh[256];
  const int t = blockIdx.x;                 // token in [0, NSROWS)
  const int n = t / SSQ, pos = t % SSQ;
  const int d = threadIdx.x;
  float* out = tok + (long long)t * DM;
  if (pos == 0) { out[d] = cls[d]; return; }
  const int kx = pos - 1;
  const float e = expr[(long long)n * KG + kx];
  float x = e * val_w[d] + val_b[d];
  sh[d] = x; __syncthreads();
  for (int s = 128; s > 0; s >>= 1) { if (d < s) sh[d] += sh[d + s]; __syncthreads(); }
  float mu = sh[0] * (1.0f / DM); __syncthreads();
  float xm = x - mu;
  sh[d] = xm * xm; __syncthreads();
  for (int s = 128; s > 0; s >>= 1) { if (d < s) sh[d] += sh[d + s]; __syncthreads(); }
  float var = sh[0] * (1.0f / DM);
  float val = gelu_exact(xm * rsqrtf(var + 1e-5f) * lg[d] + lb[d]);
  out[d] = gene_emb[(long long)kx * DM + d] + pos_emb[(long long)kx * DM + d] + val;
}

// ---------------- LayerNorm over D=256, one block per row -------------------
__global__ __launch_bounds__(256) void k_ln(const float* __restrict__ x,
                                            const float* __restrict__ g,
                                            const float* __restrict__ b,
                                            float* __restrict__ y) {
  __shared__ float sh[256];
  const long long row = blockIdx.x;
  const int d = threadIdx.x;
  float v = x[row * DM + d];
  sh[d] = v; __syncthreads();
  for (int s = 128; s > 0; s >>= 1) { if (d < s) sh[d] += sh[d + s]; __syncthreads(); }
  float mu = sh[0] * (1.0f / DM); __syncthreads();
  float xm = v - mu;
  sh[d] = xm * xm; __syncthreads();
  for (int s = 128; s > 0; s >>= 1) { if (d < s) sh[d] += sh[d + s]; __syncthreads(); }
  float var = sh[0] * (1.0f / DM);
  y[row * DM + d] = xm * rsqrtf(var + 1e-5f) * g[d] + b[d];
}

// diag[R=(b,h,n)] = 0.5*sum_d (x*64^-.25)^2 over the head slice
__global__ __launch_bounds__(256) void k_diag(const float* __restrict__ x, float* diag) {
  const int R = blockIdx.x * 8 + (threadIdx.x >> 5);
  const int lane = threadIdx.x & 31;
  if (R >= NB * HH * SSQ) return;
  const int b = R / (HH * SSQ);
  const int rem = R % (HH * SSQ);
  const int h = rem / SSQ, n = rem % SSQ;
  const float* p = x + ((long long)(b * SSQ + n)) * DM + h * DH;
  const float sc = 0.35355339059327373f;    // 64^-0.25
  float v0 = p[lane] * sc, v1 = p[lane + 32] * sc;
  float s = waveRedSum(v0 * v0 + v1 * v1);
  if (lane == 0) diag[R] = 0.5f * s;
}

__global__ __launch_bounds__(256) void k_rowmax(const float* __restrict__ dash, float* stab) {
  const int R = blockIdx.x * 8 + (threadIdx.x >> 5);
  const int lane = threadIdx.x & 31;
  if (R >= NB * HH * SSQ) return;
  const float* p = dash + (long long)R * MP;
  float mx = -3.4e38f;
  for (int j = lane; j < MF; j += 32) mx = fmaxf(mx, p[j]);
  mx = waveRedMax(mx);
  if (lane == 0) stab[R] = mx;
}

__global__ __launch_bounds__(256) void k_headmax(const float* __restrict__ dash, float* stab) {
  __shared__ float sh[256];
  const int z = blockIdx.x;                 // (b,h)
  const float* p = dash + (long long)z * SSQ * MP;
  float mx = -3.4e38f;
  for (long long e = threadIdx.x; e < (long long)SSQ * MP; e += 256) {
    int j = (int)(e % MP);
    if (j < MF) mx = fmaxf(mx, p[e]);
  }
  sh[threadIdx.x] = mx; __syncthreads();
  for (int s = 128; s > 0; s >>= 1) {
    if (threadIdx.x < s) sh[threadIdx.x] = fmaxf(sh[threadIdx.x], sh[threadIdx.x + s]);
    __syncthreads();
  }
  if (threadIdx.x == 0) stab[z] = sh[0];
}

// qp/kp = m^-0.5*(exp(dash - diag - stab)+1e-4); zero the padded feature cols
__global__ void k_expify(float* buf, const float* __restrict__ diag,
                         const float* __restrict__ stab, int perRowStab) {
  long long e = (long long)blockIdx.x * 256 + threadIdx.x;
  const long long total = (long long)NB * HH * SSQ * MP;
  if (e >= total) return;
  const int j = (int)(e % MP);
  const long long R = e / MP;
  if (j >= MF) { buf[e] = 0.f; return; }
  const float st = perRowStab ? stab[R] : stab[R / SSQ];
  const float rsm = 0.06131393394849658f;   // 266^-0.5
  buf[e] = rsm * (expf(buf[e] - diag[R] - st) + 1e-4f);
}

__global__ void k_ksum(const float* __restrict__ kp, float* ksum) {
  const int idx = blockIdx.x * 256 + threadIdx.x;
  if (idx >= NB * HH * MP) return;
  const int z = idx / MP, j = idx % MP;
  const float* p = kp + (long long)z * SSQ * MP + j;
  float s = 0.f;
  for (int n = 0; n < SSQ; ++n) s += p[(long long)n * MP];
  ksum[idx] = s;
}

__global__ __launch_bounds__(256) void k_dinv(const float* __restrict__ qp,
                                              const float* __restrict__ ksum,
                                              float* dinv) {
  const int R = blockIdx.x * 8 + (threadIdx.x >> 5);
  const int lane = threadIdx.x & 31;
  if (R >= NB * HH * SSQ) return;
  const int z = R / SSQ;
  const float* a = qp + (long long)R * MP;
  const float* b = ksum + (long long)z * MP;
  float s = 0.f;
  for (int j = lane; j < MP; j += 32) s += a[j] * b[j];
  s = waveRedSum(s);
  if (lane == 0) dinv[R] = 1.f / s;
}

// ---------------- Pooling head ----------------------------------------------
__global__ __launch_bounds__(256) void k_plogits(const float* __restrict__ q1,
                                                 const float* __restrict__ kk,
                                                 float* logit) {
  const int idx = blockIdx.x * 8 + (threadIdx.x >> 5);
  const int lane = threadIdx.x & 31;
  if (idx >= NB * KG) return;
  const int b = idx / KG;
  const float* qa = q1 + (long long)b * DM;
  const float* ka = kk + (long long)idx * DM;
  float s = 0.f;
  for (int d = lane; d < DM; d += 32) s += qa[d] * ka[d];
  s = waveRedSum(s);
  if (lane == 0) logit[idx] = s * 0.0625f;  // / sqrt(256)
}

__global__ __launch_bounds__(256) void k_psoftmax(const float* __restrict__ logit,
                                                  float* attw) {
  __shared__ float sh[256];
  const int b = blockIdx.x, t = threadIdx.x;
  const float* p = logit + (long long)b * KG;
  float mx = -3.4e38f;
  for (int g2 = t; g2 < KG; g2 += 256) mx = fmaxf(mx, p[g2]);
  sh[t] = mx; __syncthreads();
  for (int s = 128; s > 0; s >>= 1) { if (t < s) sh[t] = fmaxf(sh[t], sh[t + s]); __syncthreads(); }
  mx = sh[0]; __syncthreads();
  float sum = 0.f;
  for (int g2 = t; g2 < KG; g2 += 256) sum += expf(p[g2] - mx);
  sh[t] = sum; __syncthreads();
  for (int s = 128; s > 0; s >>= 1) { if (t < s) sh[t] += sh[t + s]; __syncthreads(); }
  const float inv = 1.f / sh[0];
  float* o = attw + (long long)b * KG;
  for (int g2 = t; g2 < KG; g2 += 256) o[g2] = expf(p[g2] - mx) * inv;
}

__global__ void k_pooled(const float* __restrict__ attw, const float* __restrict__ vv,
                         float* pooled) {
  const int idx = blockIdx.x * 256 + threadIdx.x;
  if (idx >= NB * DM) return;
  const int b = idx / DM, d = idx % DM;
  const float* w = attw + (long long)b * KG;
  const float* vp = vv + (long long)b * KG * DM + d;
  float s = 0.f;
  for (int g2 = 0; g2 < KG; ++g2) s += w[g2] * vp[(long long)g2 * DM];
  pooled[idx] = s;
}

// ---------------- Host-side launcher ----------------------------------------
static inline void launch_gemm(hipStream_t st, int M, int N, int K,
    const float* A, long long lda, int tA, long long sA1, long long sA2,
    const float* B, long long ldb, int tB, long long sB1, long long sB2,
    const float* bias, float* C, long long ldc, long long sC1, long long sC2,
    int batches, int batchDiv, float alpha, int act, int accum,
    const float* rowScale, long long sRS) {
  GemmArgs p;
  p.A = A; p.B = B; p.bias = bias; p.C = C; p.rowScale = rowScale;
  p.lda = lda; p.ldb = ldb; p.ldc = ldc;
  p.sA1 = sA1; p.sA2 = sA2; p.sB1 = sB1; p.sB2 = sB2; p.sC1 = sC1; p.sC2 = sC2;
  p.sRS = sRS; p.M = M; p.N = N; p.K = K;
  p.batchDiv = batchDiv < 1 ? 1 : batchDiv;
  p.transA = tA; p.transB = tB; p.act = act; p.accum = accum; p.alpha = alpha;
  dim3 grid((N + 63) / 64, (M + 63) / 64, batches);
  k_gemm<<<grid, dim3(256), 0, st>>>(p);
}

extern "C" void kernel_launch(void* const* d_in, const int* in_sizes, int n_in,
                              void* d_out, int out_size, void* d_ws, size_t ws_size,
                              hipStream_t stream) {
  const float* expr     = (const float*)d_in[0];
  const float* gene_emb = (const float*)d_in[1];
  const float* pos_emb  = (const float*)d_in[2];
  const float* val_w    = (const float*)d_in[3];
  const float* val_b    = (const float*)d_in[4];
  const float* val_ln_g = (const float*)d_in[5];
  const float* val_ln_b = (const float*)d_in[6];
  const float* cls      = (const float*)d_in[7];
  const float* ln1_g    = (const float*)d_in[8];
  const float* ln1_b    = (const float*)d_in[9];
  const float* wq       = (const float*)d_in[10];
  const float* bq       = (const float*)d_in[11];
  const float* wk       = (const float*)d_in[12];
  const float* bk       = (const float*)d_in[13];
  const float* wv       = (const float*)d_in[14];
  const float* bv       = (const float*)d_in[15];
  const float* wo       = (const float*)d_in[16];
  const float* bo       = (const float*)d_in[17];
  const float* proj     = (const float*)d_in[18];
  const float* ln2_g    = (const float*)d_in[19];
  const float* ln2_b    = (const float*)d_in[20];
  const float* ff_w1    = (const float*)d_in[21];
  const float* ff_b1    = (const float*)d_in[22];
  const float* ff_w2    = (const float*)d_in[23];
  const float* ff_b2    = (const float*)d_in[24];
  const float* pq_w     = (const float*)d_in[25];
  const float* pq_b     = (const float*)d_in[26];
  const float* pk_w     = (const float*)d_in[27];
  const float* pk_b     = (const float*)d_in[28];
  const float* pv_w     = (const float*)d_in[29];
  const float* pv_b     = (const float*)d_in[30];
  const float* po_w     = (const float*)d_in[31];
  const float* po_b     = (const float*)d_in[32];
  (void)in_sizes; (void)n_in; (void)out_size; (void)ws_size;

  float* ws = (float*)d_ws;
  const long long szTok = (long long)NSROWS * DM;             // 16.39M
  const long long szQP  = (long long)NB * HH * SSQ * MP;      // 69.67M

  float* tok  = ws;
  float* hbuf = tok + szTok;
  float* q    = hbuf + szTok;
  float* k    = q + szTok;
  float* v    = k + szTok;
  float* o    = v + szTok;
  float* qp   = o + szTok;
  float* kp   = qp + szQP;
  float* ffh  = qp;                                           // reuse (FF phase)
  float* ctx  = kp + szQP;                                    // NB*HH*MP*DH
  float* ksum = ctx + (long long)NB * HH * MP * DH;
  float* diag = ksum + (long long)NB * HH * MP;
  float* stabq= diag + (long long)NB * HH * SSQ;
  float* stabk= stabq + (long long)NB * HH * SSQ;
  float* dinv = stabk + (long long)NB * HH;
  float* q1   = dinv + (long long)NB * HH * SSQ;
  float* logit= q1 + (long long)NB * DM;
  float* attw = logit + (long long)NB * KG;
  float* pooled = attw + (long long)NB * KG;
  float* kk = q;                                              // reuse (pooling)
  float* vv = k;                                              // reuse (pooling)

  const int NHS = NB * HH * SSQ;
  const float xscale = 0.35355339059327373f;                  // 64^-0.25

  k_embed<<<NSROWS, 256, 0, stream>>>(expr, gene_emb, pos_emb, val_w, val_b,
                                      val_ln_g, val_ln_b, cls, tok);

  for (int l = 0; l < LLAYERS; ++l) {
    k_ln<<<NSROWS, 256, 0, stream>>>(tok, ln1_g + l * DM, ln1_b + l * DM, hbuf);

    launch_gemm(stream, NSROWS, DM, DM, hbuf, DM, 0, 0, 0,
                wq + (long long)l * DM * DM, DM, 0, 0, 0,
                bq + l * DM, q, DM, 0, 0, 1, 1, 1.f, 0, 0, nullptr, 0);
    launch_gemm(stream, NSROWS, DM, DM, hbuf, DM, 0, 0, 0,
                wk + (long long)l * DM * DM, DM, 0, 0, 0,
                bk + l * DM, k, DM, 0, 0, 1, 1, 1.f, 0, 0, nullptr, 0);
    launch_gemm(stream, NSROWS, DM, DM, hbuf, DM, 0, 0, 0,
                wv + (long long)l * DM * DM, DM, 0, 0, 0,
                bv + l * DM, v, DM, 0, 0, 1, 1, 1.f, 0, 0, nullptr, 0);

    // --- q features: dash = (q*64^-.25) @ proj^T, per (b,h) batch ---
    k_diag<<<(NHS + 7) / 8, 256, 0, stream>>>(q, diag);
    launch_gemm(stream, SSQ, MF, DH,
                q, DM, 0, (long long)SSQ * DM, DH,
                proj + (long long)l * MF * DH, DH, 1, 0, 0,
                nullptr, qp, MP, (long long)HH * SSQ * MP, (long long)SSQ * MP,
                NB * HH, HH, xscale, 0, 0, nullptr, 0);
    k_rowmax<<<(NHS + 7) / 8, 256, 0, stream>>>(qp, stabq);
    k_expify<<<(unsigned)((szQP + 255) / 256), 256, 0, stream>>>(qp, diag, stabq, 1);

    // --- k features ---
    k_diag<<<(NHS + 7) / 8, 256, 0, stream>>>(k, diag);
    launch_gemm(stream, SSQ, MF, DH,
                k, DM, 0, (long long)SSQ * DM, DH,
                proj + (long long)l * MF * DH, DH, 1, 0, 0,
                nullptr, kp, MP, (long long)HH * SSQ * MP, (long long)SSQ * MP,
                NB * HH, HH, xscale, 0, 0, nullptr, 0);
    k_headmax<<<NB * HH, 256, 0, stream>>>(kp, stabk);
    k_expify<<<(unsigned)((szQP + 255) / 256), 256, 0, stream>>>(kp, diag, stabk, 0);

    // --- linear attention ---
    k_ksum<<<(NB * HH * MP + 255) / 256, 256, 0, stream>>>(kp, ksum);
    // ctx[z] = kp[z]^T (MP x S) @ v[z] (S x DH)
    launch_gemm(stream, MP, DH, SSQ,
                kp, MP, 1, (long long)HH * SSQ * MP, (long long)SSQ * MP,
                v, DM, 0, (long long)SSQ * DM, DH,
                nullptr, ctx, DH, (long long)HH * MP * DH, (long long)MP * DH,
                NB * HH, HH, 1.f, 0, 0, nullptr, 0);
    k_dinv<<<(NHS + 7) / 8, 256, 0, stream>>>(qp, ksum, dinv);
    // o[z] = (qp[z] @ ctx[z]) * dinv, written into (b,n,h) slices of o
    launch_gemm(stream, SSQ, DH, MP,
                qp, MP, 0, (long long)HH * SSQ * MP, (long long)SSQ * MP,
                ctx, DH, 0, (long long)HH * MP * DH, (long long)MP * DH,
                nullptr, o, DM, (long long)SSQ * DM, DH,
                NB * HH, HH, 1.f, 0, 0, dinv, SSQ);
    // tok += o @ wo + bo
    launch_gemm(stream, NSROWS, DM, DM, o, DM, 0, 0, 0,
                wo + (long long)l * DM * DM, DM, 0, 0, 0,
                bo + l * DM, tok, DM, 0, 0, 1, 1, 1.f, 0, 1, nullptr, 0);

    // --- FFN ---
    k_ln<<<NSROWS, 256, 0, stream>>>(tok, ln2_g + l * DM, ln2_b + l * DM, hbuf);
    launch_gemm(stream, NSROWS, DFF, DM, hbuf, DM, 0, 0, 0,
                ff_w1 + (long long)l * DM * DFF, DFF, 0, 0, 0,
                ff_b1 + l * DFF, ffh, DFF, 0, 0, 1, 1, 1.f, 1, 0, nullptr, 0);
    launch_gemm(stream, NSROWS, DM, DFF, ffh, DFF, 0, 0, 0,
                ff_w2 + (long long)l * DFF * DM, DM, 0, 0, 0,
                ff_b2 + l * DM, tok, DM, 0, 0, 1, 1, 1.f, 0, 1, nullptr, 0);
  }

  // ---------------- pooling head ----------------
  launch_gemm(stream, NB, DM, DM, tok, (long long)SSQ * DM, 0, 0, 0,
              pq_w, DM, 0, 0, 0, pq_b, q1, DM, 0, 0, 1, 1, 1.f, 0, 0, nullptr, 0);
  launch_gemm(stream, KG, DM, DM, tok + DM, DM, 0, (long long)SSQ * DM, 0,
              pk_w, DM, 0, 0, 0, pk_b, kk, DM, (long long)KG * DM, 0,
              NB, 1, 1.f, 0, 0, nullptr, 0);
  launch_gemm(stream, KG, DM, DM, tok + DM, DM, 0, (long long)SSQ * DM, 0,
              pv_w, DM, 0, 0, 0, pv_b, vv, DM, (long long)KG * DM, 0,
              NB, 1, 1.f, 0, 0, nullptr, 0);
  k_plogits<<<(NB * KG + 7) / 8, 256, 0, stream>>>(q1, kk, logit);
  k_psoftmax<<<NB, 256, 0, stream>>>(logit, attw);
  k_pooled<<<(NB * DM + 255) / 256, 256, 0, stream>>>(attw, vv, pooled);
  launch_gemm(stream, NB, DM, DM, pooled, DM, 0, 0, 0,
              po_w, DM, 0, 0, 0, po_b, (float*)d_out, DM, 0, 0,
              1, 1, 1.f, 0, 0, nullptr, 0);
}
